// AWDLSTM_73272142069835
// MI455X (gfx1250) — matrix operand order repaired and checked by reading, not compiled, not run on hardware
//
#include <hip/hip_runtime.h>

// ---------------------------------------------------------------- types ----
typedef __attribute__((ext_vector_type(16))) __bf16 v16bf;
typedef __attribute__((ext_vector_type(8)))  __bf16 v8bf;
typedef __attribute__((ext_vector_type(8)))  float  v8f;

#define TT      70
#define BBATCH  64
#define TB      (TT*BBATCH)      // 4480 rows
#define VOCAB   33278
#define VOCABP  33280            // padded to 16
#define EMB     400
#define EMBP    416              // padded to 32
#define NHID    1150
#define HPAD    1152             // padded to 32

// ------------------------------------------------------- WMMA primitives ---
__device__ __forceinline__ v8f wmma_bf16(v16bf a, v16bf b, v8f c) {
  // (neg_a, A, neg_b, B, c_mod, C, reuse_a, reuse_b)
  return __builtin_amdgcn_wmma_f32_16x16x32_bf16(false, a, false, b, (short)0, c,
                                                 false, false);
}

// A fragment: 16x32 bf16 tile, CDNA5 layout:
//  lanes 0-15 : row M=lane, K chunks [k0..k0+7] and [k0+16..k0+23]
//  lanes 16-31: row M=lane-16, K chunks [k0+8..k0+15] and [k0+24..k0+31]
__device__ __forceinline__ v16bf load_a_frag(const __bf16* As, int ldsk, int k0,
                                             int lane) {
  int m  = lane & 15;
  int hi = lane >> 4;                 // 0 or 1
  const __bf16* p = As + m * ldsk + k0 + hi * 8;
  union { v16bf v; v8bf h[2]; } u;
  u.h[0] = *(const v8bf*)(p);         // 16B
  u.h[1] = *(const v8bf*)(p + 16);    // 16B, +32B offset
  return u.v;
}

// B fragment: 32x16 bf16 tile from row-major W[N,K]:
//  lanes 0-15 : col N=n0+lane, K chunks [k0..k0+7] and [k0+16..k0+23]
//  lanes 16-31: col N=n0+lane-16, K chunks [k0+8..k0+15] and [k0+24..k0+31]
__device__ __forceinline__ v16bf load_b_frag(const __bf16* W, int ldb, int n0,
                                             int k0, int lane) {
  int n  = n0 + (lane & 15);
  int hi = lane >> 4;
  const __bf16* p = W + (size_t)n * ldb + k0 + hi * 8;
  union { v16bf v; v8bf h[2]; } u;
  u.h[0] = *(const v8bf*)(p);
  u.h[1] = *(const v8bf*)(p + 16);
  return u.v;
}

// ------------------------------------------------ 64x32 panel per wave -----
// acc[m][s] += (A rows m*16..) x (W cols n0+s*16..). B double-buffered from
// global; A frags re-read from LDS each k-step (cheap, keeps VGPRs low).
__device__ __forceinline__ void panel_compute(const __bf16* As, int ldsk,
                                              const __bf16* __restrict__ W,
                                              int Kp, int n0, int lane,
                                              v8f (&acc)[4][2]) {
  v16bf b0 = load_b_frag(W, Kp, n0,      0, lane);
  v16bf b1 = load_b_frag(W, Kp, n0 + 16, 0, lane);
  for (int k0 = 0; k0 < Kp; k0 += 32) {
    v16bf a0 = load_a_frag(As,                ldsk, k0, lane);
    v16bf a1 = load_a_frag(As + 16 * ldsk,    ldsk, k0, lane);
    v16bf a2 = load_a_frag(As + 32 * ldsk,    ldsk, k0, lane);
    v16bf a3 = load_a_frag(As + 48 * ldsk,    ldsk, k0, lane);
    v16bf nb0, nb1;
    int k1 = k0 + 32;
    if (k1 < Kp) {
      nb0 = load_b_frag(W, Kp, n0,      k1, lane);
      nb1 = load_b_frag(W, Kp, n0 + 16, k1, lane);
    }
    acc[0][0] = wmma_bf16(a0, b0, acc[0][0]);
    acc[1][0] = wmma_bf16(a1, b0, acc[1][0]);
    acc[2][0] = wmma_bf16(a2, b0, acc[2][0]);
    acc[3][0] = wmma_bf16(a3, b0, acc[3][0]);
    acc[0][1] = wmma_bf16(a0, b1, acc[0][1]);
    acc[1][1] = wmma_bf16(a1, b1, acc[1][1]);
    acc[2][1] = wmma_bf16(a2, b1, acc[2][1]);
    acc[3][1] = wmma_bf16(a3, b1, acc[3][1]);
    b0 = nb0; b1 = nb1;
  }
}

// ------------------------------------------------------------ pack / misc ---
__global__ void pack_kernel(const float* __restrict__ src, __bf16* __restrict__ dst,
                            long long n, int Cp, int C, int gateW, int gateWp) {
  long long idx = (long long)blockIdx.x * blockDim.x + threadIdx.x;
  if (idx >= n) return;
  int rp = (int)(idx / Cp);
  int c  = (int)(idx % Cp);
  int g = rp / gateWp, j = rp % gateWp;
  float v = 0.f;
  if (j < gateW && c < C) v = src[((size_t)g * gateW + j) * C + c];
  dst[idx] = (__bf16)v;
}

__global__ void embed_kernel(const int* __restrict__ ids,
                             const float* __restrict__ embW,
                             __bf16* __restrict__ x, long long n) {
  long long idx = (long long)blockIdx.x * blockDim.x + threadIdx.x;
  if (idx >= n) return;
  int row = (int)(idx / EMBP);
  int k   = (int)(idx % EMBP);
  float v = 0.f;
  if (k < EMB) v = embW[(size_t)ids[row] * EMB + k];
  x[idx] = (__bf16)v;
}

__global__ void init_state_kernel(const float* __restrict__ hx,
                                  const float* __restrict__ cx,
                                  float* __restrict__ hst, float* __restrict__ cst,
                                  __bf16* __restrict__ hbf, int H, int HPk) {
  int idx = blockIdx.x * blockDim.x + threadIdx.x;
  if (idx >= BBATCH * HPk) return;
  int b = idx / HPk, j = idx % HPk;
  float h = 0.f, c = 0.f;
  if (j < H) { h = hx[b * H + j]; c = cx[b * H + j]; }
  hst[idx] = h; cst[idx] = c; hbf[idx] = (__bf16)h;
}

__global__ void copy_state_kernel(const float* __restrict__ hst,
                                  const float* __restrict__ cst,
                                  float* __restrict__ oh, float* __restrict__ oc,
                                  int H, int HPk) {
  int idx = blockIdx.x * blockDim.x + threadIdx.x;
  if (idx >= BBATCH * H) return;
  int b = idx / H, j = idx % H;
  oh[idx] = hst[b * HPk + j];
  oc[idx] = cst[b * HPk + j];
}

// --------------------------------------------------------------- GEMMs -----
// C[row, col](f32, ldc) = A[row,Kp](bf16) * W[col,Kp]^T + bias(gate-mapped)
// Block: 64 A rows staged in LDS; 8 waves x (4 Mtiles x 2 Ntiles) each.
template <int KMAX>
__global__ __launch_bounds__(256, 1) void gemm_bias_kernel(
    const __bf16* __restrict__ A, const __bf16* __restrict__ W,
    const float* __restrict__ bias, float* __restrict__ C,
    int Kp, int Np, int ldc, int colLimit, int gateW, int gateWp, int Nb) {
  __shared__ __bf16 As[64 * (KMAX + 8)];
  const int ldsk = Kp + 8;
  int mb = blockIdx.x / Nb;
  int nb = blockIdx.x % Nb;
  size_t row0 = (size_t)mb * 64;

  int chunks = Kp / 8;                     // 16B chunks per row
  int total = 64 * chunks;
  for (int i = threadIdx.x; i < total; i += blockDim.x) {
    int r = i / chunks, c = i % chunks;
    ((uint4*)(As + r * ldsk))[c] = ((const uint4*)(A + (row0 + r) * Kp))[c];
  }
  __syncthreads();

  int w = threadIdx.x >> 5, lane = threadIdx.x & 31;
  int n0 = (nb * 8 + w) * 32;
  if (n0 >= Np) return;

  v8f acc[4][2] = {};
  panel_compute(As, ldsk, W, Kp, n0, lane, acc);

#pragma unroll
  for (int m = 0; m < 4; ++m) {
    int rbase = (int)row0 + m * 16 + ((lane >> 4) << 3);
#pragma unroll
    for (int s = 0; s < 2; ++s) {
      int col = n0 + s * 16 + (lane & 15);
      if (col < colLimit) {
        int g = col / gateWp, j = col % gateWp;
        float bv = (j < gateW) ? bias[g * gateW + j] : 0.f;
#pragma unroll
        for (int r = 0; r < 8; ++r)
          C[(size_t)(rbase + r) * ldc + col] = acc[m][s][r] + bv;
      }
    }
  }
}

// gates[64,Np] = Hbf[64,Kp] * Whh^T + gi_t  (recurrent step)
template <int KMAX>
__global__ __launch_bounds__(256, 1) void lstm_gemm_kernel(
    const __bf16* __restrict__ Hbf, const __bf16* __restrict__ Whh,
    const float* __restrict__ gi_t, float* __restrict__ gates, int Kp, int Np) {
  __shared__ __bf16 As[64 * (KMAX + 8)];
  const int ldsk = Kp + 8;

  int chunks = Kp / 8;
  int total = 64 * chunks;
  for (int i = threadIdx.x; i < total; i += blockDim.x) {
    int r = i / chunks, c = i % chunks;
    ((uint4*)(As + r * ldsk))[c] = ((const uint4*)(Hbf + (size_t)r * Kp))[c];
  }
  __syncthreads();

  int w = threadIdx.x >> 5, lane = threadIdx.x & 31;
  int n0 = (blockIdx.x * 8 + w) * 32;
  if (n0 >= Np) return;

  v8f acc[4][2] = {};
  panel_compute(As, ldsk, Whh, Kp, n0, lane, acc);

#pragma unroll
  for (int m = 0; m < 4; ++m) {
    int rbase = m * 16 + ((lane >> 4) << 3);
#pragma unroll
    for (int s = 0; s < 2; ++s) {
      int col = n0 + s * 16 + (lane & 15);
#pragma unroll
      for (int r = 0; r < 8; ++r) {
        size_t o = (size_t)(rbase + r) * Np + col;
        gates[o] = acc[m][s][r] + gi_t[o];
      }
    }
  }
}

// LSTM pointwise cell
__global__ void lstm_pointwise_kernel(const float* __restrict__ gates,
                                      float* __restrict__ cst, float* __restrict__ hst,
                                      __bf16* __restrict__ hbf, __bf16* __restrict__ ybf_t,
                                      int H, int HPk, int Np) {
  int idx = blockIdx.x * blockDim.x + threadIdx.x;
  if (idx >= BBATCH * HPk) return;
  int b = idx / HPk, j = idx % HPk;
  if (j < H) {
    size_t ro = (size_t)b * Np;
    float ig = gates[ro + 0 * HPk + j];
    float fg = gates[ro + 1 * HPk + j];
    float gg = gates[ro + 2 * HPk + j];
    float og = gates[ro + 3 * HPk + j];
    ig = 1.f / (1.f + __expf(-ig));
    fg = 1.f / (1.f + __expf(-fg));
    gg = tanhf(gg);
    og = 1.f / (1.f + __expf(-og));
    float c = fg * cst[idx] + ig * gg;
    float h = og * tanhf(c);
    cst[idx] = c; hst[idx] = h;
    __bf16 hb = (__bf16)h;
    hbf[idx] = hb; ybf_t[idx] = hb;
  } else {
    hbf[idx] = (__bf16)0.f; ybf_t[idx] = (__bf16)0.f;
  }
}

// ------------------------------------------------------------- launcher ----
extern "C" void kernel_launch(void* const* d_in, const int* in_sizes, int n_in,
                              void* d_out, int out_size, void* d_ws, size_t ws_size,
                              hipStream_t stream) {
  (void)in_sizes; (void)n_in; (void)out_size; (void)ws_size;

  const int*   ids   = (const int*)  d_in[0];
  const float* embW  = (const float*)d_in[1];
  const float* wih0  = (const float*)d_in[2];
  const float* whh0  = (const float*)d_in[3];
  const float* b0    = (const float*)d_in[4];
  const float* wih1  = (const float*)d_in[5];
  const float* whh1  = (const float*)d_in[6];
  const float* b1    = (const float*)d_in[7];
  const float* wih2  = (const float*)d_in[8];
  const float* whh2  = (const float*)d_in[9];
  const float* b2    = (const float*)d_in[10];
  const float* decb  = (const float*)d_in[11];
  const float* hx0   = (const float*)d_in[12];
  const float* cx0   = (const float*)d_in[13];
  const float* hx1   = (const float*)d_in[14];
  const float* cx1   = (const float*)d_in[15];
  const float* hx2   = (const float*)d_in[16];
  const float* cx2   = (const float*)d_in[17];

  char* p = (char*)d_ws;
  auto carve = [&](size_t bytes) -> char* {
    char* r = p; p += (bytes + 255) & ~(size_t)255; return r;
  };
  __bf16* Wih0p = (__bf16*)carve((size_t)4608 * EMBP * 2);
  __bf16* Whh0p = (__bf16*)carve((size_t)4608 * HPAD * 2);
  __bf16* Wih1p = (__bf16*)carve((size_t)4608 * HPAD * 2);
  __bf16* Whh1p = (__bf16*)carve((size_t)4608 * HPAD * 2);
  __bf16* Wih2p = (__bf16*)carve((size_t)1664 * HPAD * 2);
  __bf16* Whh2p = (__bf16*)carve((size_t)1664 * EMBP * 2);
  __bf16* EmbWp = (__bf16*)carve((size_t)VOCABP * EMBP * 2);
  __bf16* Xemb  = (__bf16*)carve((size_t)TB * EMBP * 2);
  __bf16* Y0    = (__bf16*)carve((size_t)TB * HPAD * 2);
  __bf16* Y1    = (__bf16*)carve((size_t)TB * HPAD * 2);
  __bf16* Y2    = (__bf16*)carve((size_t)TB * EMBP * 2);
  float*  gi    = (float*) carve((size_t)TB * 4608 * 4);
  float*  gates = (float*) carve((size_t)BBATCH * 4608 * 4);
  float*  hst   = (float*) carve((size_t)BBATCH * HPAD * 4);
  float*  cst   = (float*) carve((size_t)BBATCH * HPAD * 4);
  __bf16* hbf   = (__bf16*)carve((size_t)BBATCH * HPAD * 2);

  auto packL = [&](const float* src, __bf16* dst, long long Rp, int Cp, int C,
                   int gW, int gWp) {
    long long n = Rp * Cp;
    pack_kernel<<<(unsigned)((n + 255) / 256), 256, 0, stream>>>(src, dst, n, Cp, C, gW, gWp);
  };
  packL(wih0, Wih0p, 4608, EMBP, EMB,  NHID, HPAD);
  packL(whh0, Whh0p, 4608, HPAD, NHID, NHID, HPAD);
  packL(wih1, Wih1p, 4608, HPAD, NHID, NHID, HPAD);
  packL(whh1, Whh1p, 4608, HPAD, NHID, NHID, HPAD);
  packL(wih2, Wih2p, 1664, HPAD, NHID, EMB,  EMBP);
  packL(whh2, Whh2p, 1664, EMBP, EMB,  EMB,  EMBP);
  packL(embW, EmbWp, VOCABP, EMBP, EMB, VOCAB, VOCABP);

  {
    long long n = (long long)TB * EMBP;
    embed_kernel<<<(unsigned)((n + 255) / 256), 256, 0, stream>>>(ids, embW, Xemb, n);
  }

  float* out = (float*)d_out;
  float* dec = out;
  float* oh0 = out + (size_t)TB * VOCAB;
  float* oc0 = oh0 + (size_t)BBATCH * NHID;
  float* oh1 = oc0 + (size_t)BBATCH * NHID;
  float* oc1 = oh1 + (size_t)BBATCH * NHID;
  float* oh2 = oc1 + (size_t)BBATCH * NHID;
  float* oc2 = oh2 + (size_t)BBATCH * EMB;

  auto run_layer = [&](const __bf16* xin, int Kin, const __bf16* Wihp,
                       const __bf16* Whhp, const float* bias, const float* hx,
                       const float* cx, int H, int HPk, __bf16* ybf,
                       float* out_h, float* out_c) {
    int Np = 4 * HPk;
    int Nb = (Np + 255) / 256;               // 256 output cols per block
    // gi = xin * Wih^T + b  (all timesteps batched)
    if (Kin == EMBP)
      gemm_bias_kernel<EMBP><<<(TB / 64) * Nb, 256, 0, stream>>>(
          xin, Wihp, bias, gi, Kin, Np, Np, Np, H, HPk, Nb);
    else
      gemm_bias_kernel<HPAD><<<(TB / 64) * Nb, 256, 0, stream>>>(
          xin, Wihp, bias, gi, Kin, Np, Np, Np, H, HPk, Nb);

    init_state_kernel<<<(BBATCH * HPk + 255) / 256, 256, 0, stream>>>(
        hx, cx, hst, cst, hbf, H, HPk);

    int pwGrid = (BBATCH * HPk + 255) / 256;
    if (HPk == HPAD) {
      for (int t = 0; t < TT; ++t) {
        lstm_gemm_kernel<HPAD><<<Nb, 256, 0, stream>>>(
            hbf, Whhp, gi + (size_t)t * BBATCH * Np, gates, HPk, Np);
        lstm_pointwise_kernel<<<pwGrid, 256, 0, stream>>>(
            gates, cst, hst, hbf, ybf + (size_t)t * BBATCH * HPk, H, HPk, Np);
      }
    } else {
      for (int t = 0; t < TT; ++t) {
        lstm_gemm_kernel<EMBP><<<Nb, 256, 0, stream>>>(
            hbf, Whhp, gi + (size_t)t * BBATCH * Np, gates, HPk, Np);
        lstm_pointwise_kernel<<<pwGrid, 256, 0, stream>>>(
            gates, cst, hst, hbf, ybf + (size_t)t * BBATCH * HPk, H, HPk, Np);
      }
    }
    copy_state_kernel<<<(BBATCH * H + 255) / 256, 256, 0, stream>>>(
        hst, cst, out_h, out_c, H, HPk);
  };

  run_layer(Xemb, EMBP, Wih0p, Whh0p, b0, hx0, cx0, NHID, HPAD, Y0, oh0, oc0);
  run_layer(Y0,   HPAD, Wih1p, Whh1p, b1, hx1, cx1, NHID, HPAD, Y1, oh1, oc1);
  run_layer(Y1,   HPAD, Wih2p, Whh2p, b2, hx2, cx2, EMB,  EMBP, Y2, oh2, oc2);

  // tied decoder: out[4480,VOCAB] = Y2 * EmbWp^T + dec_b
  {
    int Nb = VOCABP / 256;                   // 130
    gemm_bias_kernel<EMBP><<<(TB / 64) * Nb, 256, 0, stream>>>(
        Y2, EmbWp, decb, dec, EMBP, VOCABP, VOCAB, VOCAB, VOCAB, VOCABP, Nb);
  }
}
